// OptimEDM_5308579578174
// MI455X (gfx1250) — matrix operand order repaired and checked by reading, not compiled, not run on hardware
//
#include <hip/hip_runtime.h>

// ---------------------------------------------------------------------------
// EDM softmax denoiser for MI455X (gfx1250, wave32, WMMA bf16 16x16x32).
//   d2 = x2 + y2 - 2 x@y^T ; softmax over N ; out = prob @ y
// B=512, D=3072, N=50000 (padded to 50176 = 196*256).
// Both GEMMs use v_wmma_f32_16x16x32_bf16 with fp32 accumulation; norms and
// softmax stay fp32. All reductions are fixed-order (deterministic):
// gemm2 uses split-K x4 into private partials + a fixed-order reduce kernel,
// and ping-pong LDS double buffering (one barrier per 64-K chunk).
// ---------------------------------------------------------------------------

#define BB     512
#define DD     3072
#define NN     50000
#define NPAD   50176           // 196 * 256 = 392 * 128
#define KSPLIT 4
#define KCHUNK (NPAD / KSPLIT) // 12544, multiple of 64

typedef __attribute__((ext_vector_type(16))) __bf16 v16bf;
typedef __attribute__((ext_vector_type(8)))  __bf16 v8bf;
typedef __attribute__((ext_vector_type(4)))  __bf16 v4bf;
typedef __attribute__((ext_vector_type(2)))  __bf16 v2bf;
typedef __attribute__((ext_vector_type(8)))  float  v8f;

// ---- small reduction helpers (wave32) -------------------------------------
__device__ __forceinline__ float waveMax(float v) {
#pragma unroll
  for (int o = 16; o; o >>= 1) v = fmaxf(v, __shfl_xor(v, o, 32));
  return v;
}
__device__ __forceinline__ float waveSum(float v) {
#pragma unroll
  for (int o = 16; o; o >>= 1) v += __shfl_xor(v, o, 32);
  return v;
}
__device__ __forceinline__ float blockMax(float v, float* sh) {
  v = waveMax(v);
  int wave = threadIdx.x >> 5, lane = threadIdx.x & 31;
  if (lane == 0) sh[wave] = v;
  __syncthreads();
  if (wave == 0) {
    float w = (lane < 8) ? sh[lane] : -3.4e38f;
    w = waveMax(w);
    if (lane == 0) sh[0] = w;
  }
  __syncthreads();
  float r = sh[0];
  __syncthreads();
  return r;
}
__device__ __forceinline__ float blockSum(float v, float* sh) {
  v = waveSum(v);
  int wave = threadIdx.x >> 5, lane = threadIdx.x & 31;
  if (lane == 0) sh[wave] = v;
  __syncthreads();
  if (wave == 0) {
    float w = (lane < 8) ? sh[lane] : 0.0f;
    w = waveSum(w);
    if (lane == 0) sh[0] = w;
  }
  __syncthreads();
  float r = sh[0];
  __syncthreads();
  return r;
}

// A-fragment (16x32 bf16, row-major source, K contiguous):
//   lanes 0-15  : row M = lane,    K in {0..15}
//   lanes 16-31 : row M = lane-16, K in {16..31}
// p must already point at rowBase + half*16 + k.
__device__ __forceinline__ v16bf loadAfrag(const __bf16* p) {
  v8bf a0 = *(const v8bf*)p;
  v8bf a1 = *(const v8bf*)(p + 8);
  return __builtin_shufflevector(a0, a1, 0, 1, 2, 3, 4, 5, 6, 7,
                                 8, 9, 10, 11, 12, 13, 14, 15);
}
__device__ __forceinline__ v16bf load16(const __bf16* p) {
  v8bf a0 = *(const v8bf*)p;
  v8bf a1 = *(const v8bf*)(p + 8);
  return __builtin_shufflevector(a0, a1, 0, 1, 2, 3, 4, 5, 6, 7,
                                 8, 9, 10, 11, 12, 13, 14, 15);
}

// ---------------------------------------------------------------------------
// Kernel 1: fp32 -> bf16 conversion + row squared norms + 1/(2 sigma^2).
// One block per row; rows [0,512) = x, rows [512, 512+NPAD) = y (pad rows
// zeroed, pad y2 = 1e30 so softmax weight underflows to exactly 0).
// ---------------------------------------------------------------------------
__global__ __launch_bounds__(256) void prep_kernel(
    const float* __restrict__ x, const float* __restrict__ sigma,
    const float* __restrict__ y, __bf16* __restrict__ xb,
    __bf16* __restrict__ yb, float* __restrict__ x2, float* __restrict__ y2,
    float* __restrict__ inv2s2) {
  __shared__ float sh[8];
  const int r = blockIdx.x;
  const int t = threadIdx.x;

  if (r < BB) {
    const float4* src = (const float4*)(x + (size_t)r * DD);
    __bf16* dst = xb + (size_t)r * DD;
    float ss = 0.0f;
#pragma unroll
    for (int i = 0; i < 3; ++i) {
      float4 v = src[t + 256 * i];
      ss += v.x * v.x + v.y * v.y + v.z * v.z + v.w * v.w;
      v4bf b = {(__bf16)v.x, (__bf16)v.y, (__bf16)v.z, (__bf16)v.w};
      *(v4bf*)(dst + 4 * (t + 256 * i)) = b;
    }
    ss = blockSum(ss, sh);
    if (t == 0) {
      x2[r] = ss;
      float s = sigma[r];
      inv2s2[r] = 1.0f / (2.0f * s * s);
    }
  } else {
    const int n = r - BB;
    __bf16* dst = yb + (size_t)n * DD;
    if (n < NN) {
      const float4* src = (const float4*)(y + (size_t)n * DD);
      float ss = 0.0f;
#pragma unroll
      for (int i = 0; i < 3; ++i) {
        float4 v = src[t + 256 * i];
        ss += v.x * v.x + v.y * v.y + v.z * v.z + v.w * v.w;
        v4bf b = {(__bf16)v.x, (__bf16)v.y, (__bf16)v.z, (__bf16)v.w};
        *(v4bf*)(dst + 4 * (t + 256 * i)) = b;
      }
      ss = blockSum(ss, sh);
      if (t == 0) y2[n] = ss;
    } else {
#pragma unroll
      for (int i = 0; i < 3; ++i) {
        v4bf z = {};
        *(v4bf*)(dst + 4 * (t + 256 * i)) = z;
      }
      if (t == 0) y2[n] = 1e30f;  // logits -> -huge -> exp == 0
    }
  }
}

// ---------------------------------------------------------------------------
// Kernel 2: logits GEMM.  L[m][n] = -max(x2[m]+y2[n]-2*dot, 0) * inv2s2[m]
// Block tile 128(M) x 256(N); 8 waves each computing 64x64 (4x4 frags):
// 16 WMMAs per 16 b128 fragment loads per k-step. Both operands are
// K-contiguous in memory -> direct b128 fragment loads; grid.x = M-tiles so
// the y tile is L2-reused across all 4 M-tiles.
// ---------------------------------------------------------------------------
__global__ __launch_bounds__(256) void gemm1_kernel(
    const __bf16* __restrict__ xb, const __bf16* __restrict__ yb,
    const float* __restrict__ x2, const float* __restrict__ y2,
    const float* __restrict__ inv2s2, float* __restrict__ L) {
  const int lane = threadIdx.x & 31;
  const int wave = threadIdx.x >> 5;
  const int half = lane >> 4;
  const int l16 = lane & 15;
  const int wm = wave >> 2;  // 0..1
  const int wn = wave & 3;   // 0..3
  const int m0 = blockIdx.x * 128 + wm * 64;
  const int n0 = blockIdx.y * 256 + wn * 64;

  v8f acc[4][4] = {};
  const __bf16* aptr = xb + (size_t)(m0 + l16) * DD + half * 16;
  const __bf16* bptr = yb + (size_t)(n0 + l16) * DD + half * 16;

  for (int k = 0; k < DD; k += 32) {
    v16bf Bf[4];
#pragma unroll
    for (int fn = 0; fn < 4; ++fn)
      Bf[fn] = load16(bptr + (size_t)fn * 16 * DD + k);
    v16bf Af[4];
#pragma unroll
    for (int fm = 0; fm < 4; ++fm)
      Af[fm] = loadAfrag(aptr + (size_t)fm * 16 * DD + k);
#pragma unroll
    for (int fm = 0; fm < 4; ++fm)
#pragma unroll
      for (int fn = 0; fn < 4; ++fn)
        acc[fm][fn] = __builtin_amdgcn_wmma_f32_16x16x32_bf16(
            false, Af[fm], false, Bf[fn], (short)0, acc[fm][fn], false, false);
  }

#pragma unroll
  for (int fm = 0; fm < 4; ++fm) {
#pragma unroll
    for (int fn = 0; fn < 4; ++fn) {
      const int n = n0 + fn * 16 + l16;
      const float y2n = y2[n];
#pragma unroll
      for (int j = 0; j < 8; ++j) {
        const int m = m0 + fm * 16 + half * 8 + j;
        float d2 = x2[m] + y2n - 2.0f * acc[fm][fn][j];
        d2 = fmaxf(d2, 0.0f);
        L[(size_t)m * NPAD + n] = -d2 * inv2s2[m];
      }
    }
  }
}

// ---------------------------------------------------------------------------
// Kernel 3: row softmax. One block per row: pass1 max, pass2 exp (bf16) + sum.
// Pad columns have logits ~ -1e30 -> exp underflows to 0 -> uniform handling.
// ---------------------------------------------------------------------------
__global__ __launch_bounds__(256) void softmax_kernel(
    const float* __restrict__ L, __bf16* __restrict__ E,
    float* __restrict__ rowscale) {
  __shared__ float sh[8];
  const int b = blockIdx.x;
  const float* row = L + (size_t)b * NPAD;
  __bf16* erow = E + (size_t)b * NPAD;

  float mx = -3.4e38f;
  for (int n = threadIdx.x; n < NPAD; n += 256) mx = fmaxf(mx, row[n]);
  mx = blockMax(mx, sh);

  float sum = 0.0f;
  for (int n = threadIdx.x; n < NPAD; n += 256) {
    float e = __expf(row[n] - mx);
    erow[n] = (__bf16)e;
    sum += e;
  }
  sum = blockSum(sum, sh);
  if (threadIdx.x == 0) rowscale[b] = 1.0f / sum;
}

// ---------------------------------------------------------------------------
// Kernel 4: split-K GEMM.  partial[s][m][d] = sum_{n in chunk s} E[m][n]*yb[n][d]
// Block tile 128(M) x 128(d), 8 waves x (64x32), K chunk of 12544 per split
// (384 blocks -> fills the chip). B operand (yb) is d-contiguous, so stage a
// 64(K) x 128(d) tile into LDS transposed (Bt[buf][d][k], row stride 72 bf16
// = 144 B, 16B-aligned rows -> clean ds_load_b128 fragment reads).
// Ping-pong double buffering: one barrier per chunk; next chunk's global
// loads are issued into registers before the current chunk's WMMAs, hiding
// global->LDS latency. Safe with a single barrier because writes to a given
// buffer can only happen after the barrier that follows all reads of it.
// ---------------------------------------------------------------------------
__global__ __launch_bounds__(256) void gemm2_kernel(
    const __bf16* __restrict__ E, const __bf16* __restrict__ yb,
    float* __restrict__ partial) {
  __shared__ __bf16 Bt[2][128][72];  // 2 x 18 KB

  const int lane = threadIdx.x & 31;
  const int wave = threadIdx.x >> 5;
  const int half = lane >> 4;
  const int l16 = lane & 15;
  const int wm = wave >> 2;  // 0..1
  const int wn = wave & 3;   // 0..3
  const int m0 = blockIdx.x * 128 + wm * 64;
  const int d0 = blockIdx.y * 128;  // block's d base
  const int dW = wn * 32;           // wave's d offset inside tile
  const int kbase = blockIdx.z * KCHUNK;

  // staging assignment: thread -> 2 consecutive k-rows x 16 d columns
  const int kk2 = (threadIdx.x >> 3) * 2;    // 0,2,..,62
  const int dpart = (threadIdx.x & 7) * 16;  // 0..112

  v8f acc[4][2] = {};
  const __bf16* aptr = E + (size_t)(m0 + l16) * NPAD + kbase + half * 16;

  // prefetch chunk 0 into registers
  const __bf16* srow = yb + (size_t)(kbase + kk2) * DD + d0 + dpart;
  v16bf r0 = load16(srow);
  v16bf r1 = load16(srow + DD);

  int buf = 0;
  for (int k0 = 0; k0 < KCHUNK; k0 += 64) {
    // ---- commit prefetched chunk into Bt[buf] (transposed) ----
#pragma unroll
    for (int i = 0; i < 16; ++i) {
      v2bf pr = {r0[i], r1[i]};
      *(v2bf*)&Bt[buf][dpart + i][kk2] = pr;  // 4B store, kk2 even -> aligned
    }
    __syncthreads();

    // ---- prefetch next chunk from global (overlaps with WMMAs below) ----
    if (k0 + 64 < KCHUNK) {
      const __bf16* s2 = yb + (size_t)(kbase + k0 + 64 + kk2) * DD + d0 + dpart;
      r0 = load16(s2);
      r1 = load16(s2 + DD);
    }

    // ---- compute from Bt[buf] ----
#pragma unroll
    for (int ks = 0; ks < 2; ++ks) {
      const int k = k0 + ks * 32;
      v16bf Bf[2];
      Bf[0] = load16(&Bt[buf][dW + l16][ks * 32 + half * 16]);
      Bf[1] = load16(&Bt[buf][dW + 16 + l16][ks * 32 + half * 16]);
      v16bf Af[4];
#pragma unroll
      for (int fm = 0; fm < 4; ++fm)
        Af[fm] = loadAfrag(aptr + (size_t)fm * 16 * NPAD + k);
#pragma unroll
      for (int fm = 0; fm < 4; ++fm)
#pragma unroll
        for (int fn = 0; fn < 2; ++fn)
          acc[fm][fn] = __builtin_amdgcn_wmma_f32_16x16x32_bf16(
              false, Af[fm], false, Bf[fn], (short)0, acc[fm][fn], false,
              false);
    }
    buf ^= 1;
  }

  float* pout = partial + (size_t)blockIdx.z * BB * DD;
#pragma unroll
  for (int fm = 0; fm < 4; ++fm) {
#pragma unroll
    for (int fn = 0; fn < 2; ++fn) {
      const int d = d0 + dW + fn * 16 + l16;
#pragma unroll
      for (int j = 0; j < 8; ++j) {
        const int m = m0 + fm * 16 + half * 8 + j;
        pout[(size_t)m * DD + d] = acc[fm][fn][j];
      }
    }
  }
}

// ---------------------------------------------------------------------------
// Kernel 5: fixed-order split-K reduction + softmax normalization.
// ---------------------------------------------------------------------------
__global__ __launch_bounds__(256) void reduce_kernel(
    const float* __restrict__ partial, const float* __restrict__ rowscale,
    float* __restrict__ out) {
  const size_t i = (size_t)blockIdx.x * 256 + threadIdx.x;  // < BB*DD
  const int m = (int)(i / DD);
  float s = partial[i];
#pragma unroll
  for (int sp = 1; sp < KSPLIT; ++sp) s += partial[i + (size_t)sp * BB * DD];
  out[i] = s * rowscale[m];
}

// ---------------------------------------------------------------------------
extern "C" void kernel_launch(void* const* d_in, const int* in_sizes, int n_in,
                              void* d_out, int out_size, void* d_ws,
                              size_t ws_size, hipStream_t stream) {
  const float* x = (const float*)d_in[0];      // (512, 3, 32, 32) fp32
  const float* sigma = (const float*)d_in[1];  // (512,) fp32
  const float* y = (const float*)d_in[2];      // (50000, 3072) fp32
  float* out = (float*)d_out;                  // (512, 3072) fp32

  char* ws = (char*)d_ws;
  size_t off = 0;
  auto carve = [&](size_t bytes) -> void* {
    void* p = ws + off;
    off += (bytes + 255) & ~(size_t)255;
    return p;
  };
  __bf16* xb = (__bf16*)carve((size_t)BB * DD * 2);              //   3.1 MB
  __bf16* yb = (__bf16*)carve((size_t)NPAD * DD * 2);            // 308.3 MB
  float* x2 = (float*)carve((size_t)BB * 4);
  float* y2 = (float*)carve((size_t)NPAD * 4);
  float* inv2s2 = (float*)carve((size_t)BB * 4);
  float* rowscale = (float*)carve((size_t)BB * 4);
  float* L = (float*)carve((size_t)BB * NPAD * 4);               // 102.8 MB
  __bf16* E = (__bf16*)carve((size_t)BB * NPAD * 2);             //  51.4 MB
  float* partial = (float*)carve((size_t)KSPLIT * BB * DD * 4);  //  25.2 MB

  prep_kernel<<<BB + NPAD, 256, 0, stream>>>(x, sigma, y, xb, yb, x2, y2,
                                             inv2s2);
  gemm1_kernel<<<dim3(BB / 128, NPAD / 256), 256, 0, stream>>>(xb, yb, x2, y2,
                                                               inv2s2, L);
  softmax_kernel<<<BB, 256, 0, stream>>>(L, E, rowscale);
  gemm2_kernel<<<dim3(BB / 128, DD / 128, KSPLIT), 256, 0, stream>>>(E, yb,
                                                                     partial);
  reduce_kernel<<<(BB * DD) / 256, 256, 0, stream>>>(partial, rowscale, out);
}
